// Right2Left_79018808312175
// MI455X (gfx1250) — compile-verified, matrix operand order checked by reading, not run on hardware
//
#include <hip/hip_runtime.h>
#include <hip/hip_bf16.h>

// Problem constants (from reference)
#define H 256
#define SEVEN_H 1792
#define TWO_H 512
#define VOCAB 35
#define IDX_PAD 34
#define T_ 256
#define NSEQ 128          // B*P
#define SEQ_PER_WG 16
#define NWG 8             // 128 / 16
#define TSTEPS 255        // i = 1..255 (reverse scan)
#define OUT_SZ (NSEQ * TSTEPS * H)   // elements per output tensor

typedef __attribute__((ext_vector_type(16))) __bf16        v16bf;
typedef __attribute__((ext_vector_type(8)))  float         v8f;
typedef __attribute__((ext_vector_type(4)))  unsigned int  u32x4;

union ABu { u32x4 u[2]; v16bf v; };

#define XS 264    // padded row stride for h-state (bf16), 256 + 8
#define ZS 1800   // padded row stride for Z (f32), 1792 + 8

// ---------------- prep 1: recurrent weights fp32 -> bf16 ----------------
__global__ void ctlstm_prep_wh(const float* __restrict__ W, __bf16* __restrict__ wh) {
    int idx = blockIdx.x * blockDim.x + threadIdx.x;
    if (idx < SEVEN_H * H) {
        int j = idx / H, k = idx % H;
        wh[idx] = (__bf16)W[j * TWO_H + H + k];   // W[:, H:2H]
    }
}

// ---- prep 2: Proj[v, j] = b[j] + Emb[v] . W[j, 0:H]  (kills input GEMM) ----
__global__ void ctlstm_prep_proj(const float* __restrict__ Emb, const float* __restrict__ W,
                                 const float* __restrict__ b, float* __restrict__ proj) {
    int idx = blockIdx.x * blockDim.x + threadIdx.x;
    if (idx < VOCAB * SEVEN_H) {
        int v = idx / SEVEN_H, j = idx % SEVEN_H;
        float s = b[j];
        const float* e = Emb + v * H;
        const float* w = W + j * TWO_H;
        #pragma unroll 4
        for (int k = 0; k < H; ++k) s = fmaf(e[k], w[k], s);
        proj[idx] = s;
    }
}

__device__ __forceinline__ float sigf(float x) { return 1.f / (1.f + __expf(-x)); }

// ---------------- main: per-WG private 255-step reverse scan ----------------
__global__ void __launch_bounds__(256, 1) ctlstm_scan(
    const int* __restrict__ ev, const float* __restrict__ dt,
    const float* __restrict__ proj, const __bf16* __restrict__ wh,
    float* __restrict__ out) {
    __shared__ __bf16 Xsh[16 * XS];   // h_plus state, [16 seq][H] bf16 (padded)
    __shared__ float  Zsh[16 * ZS];   // z = h @ Wh^T,  [16 seq][7H] f32 (padded)

    const int tid  = threadIdx.x;
    const int g    = blockIdx.x;          // 0..7 -> sequences [16g, 16g+16)
    const int wave = tid >> 5;            // 8 waves
    const int lane = tid & 31;
    const int nloc = lane & 15;
    const int half = lane >> 4;
    const int aoff = half * 8;            // K sub-offset per lane half

    // elementwise mapping: thread -> (seq m, h = hb + 16*jj)
    const int m  = tid >> 4;
    const int hb = tid & 15;
    const int s  = g * SEQ_PER_WG + m;

    for (int idx = tid; idx < 16 * XS; idx += 256) Xsh[idx] = (__bf16)0.f;
    float c_st[16], cb_st[16];
    #pragma unroll
    for (int jj = 0; jj < 16; ++jj) { c_st[jj] = 0.f; cb_st[jj] = 0.f; }
    __syncthreads();

    for (int i = TSTEPS; i >= 1; --i) {
        // ---------- GEMM phase: Z[16,1792] = h_plus @ Wh^T (bf16 WMMA, f32 acc) ----------
        ABu a[8];
        #pragma unroll
        for (int kt = 0; kt < 8; ++kt) {          // K = 8 x 32
            const __bf16* xr = &Xsh[nloc * XS + kt * 32 + aoff];
            a[kt].u[0] = *(const u32x4*)(xr);
            a[kt].u[1] = *(const u32x4*)(xr + 16);
        }
        for (int t = 0; t < 14; ++t) {            // 8 waves x 14 = 112 N-tiles = 7H/16
            const int nt = wave * 14 + t;
            const __bf16* wr = wh + (nt * 16 + nloc) * H;
            __builtin_prefetch(wr + 16 * H, 0, 3);  // next tile's row (near-cache hint)
            // Two independent accumulator chains (even/odd k-tiles) -> 4-deep
            // dependent WMMA chains instead of 8-deep; XDL can interleave them.
            v8f acc0 = {}, acc1 = {};
            #pragma unroll
            for (int kt = 0; kt < 8; kt += 2) {
                ABu b0, b1;
                b0.u[0] = *(const u32x4*)(wr + kt * 32 + aoff);
                b0.u[1] = *(const u32x4*)(wr + kt * 32 + 16 + aoff);
                b1.u[0] = *(const u32x4*)(wr + (kt + 1) * 32 + aoff);
                b1.u[1] = *(const u32x4*)(wr + (kt + 1) * 32 + 16 + aoff);
                acc0 = __builtin_amdgcn_wmma_f32_16x16x32_bf16(
                           false, a[kt].v, false, b0.v, (short)0, acc0, false, false);
                acc1 = __builtin_amdgcn_wmma_f32_16x16x32_bf16(
                           false, a[kt + 1].v, false, b1.v, (short)0, acc1, false, false);
            }
            const v8f acc = acc0 + acc1;
            #pragma unroll
            for (int r = 0; r < 8; ++r)           // D: VGPR r -> M = r + 8*half, N = nloc
                Zsh[(r + half * 8) * ZS + nt * 16 + nloc] = acc[r];
        }
        __syncthreads();

        // ---------- elementwise recurrence ----------
        const int   e    = ev[s * T_ + i];
        const float d    = dt[s * T_ + i];
        const float mask = (e != IDX_PAD) ? 1.f : 0.f;
        const float* pr  = proj + e * SEVEN_H;
        const long  ob   = (long)s * TSTEPS * H + (long)(i - 1) * H;
        #pragma unroll
        for (int jj = 0; jj < 16; ++jj) {
            const int h = hb + 16 * jj;
            const float* zr = &Zsh[m * ZS + h];
            const float gi  = sigf(zr[0 * H] + pr[0 * H + h]);
            const float gf  = sigf(zr[1 * H] + pr[1 * H + h]);
            const float go  = sigf(zr[2 * H] + pr[2 * H + h]);
            const float gz  = tanhf(zr[3 * H] + pr[3 * H + h]);
            const float gib = sigf(zr[4 * H] + pr[4 * H + h]);
            const float gfb = sigf(zr[5 * H] + pr[5 * H + h]);
            const float gdr = zr[6 * H] + pr[6 * H + h];
            const float gd  = (gdr > 20.f) ? gdr : log1pf(__expf(gdr));  // softplus
            const float cell = gf * c_st[jj] + gi * gz;
            const float cbar = gfb * cb_st[jj] + gib * gz;
            const float hmin = go * tanhf(cell);
            const float dec  = __expf(-gd * d);
            float c_im1 = cbar + (cell - cbar) * dec;
            float h_im1 = go * tanhf(c_im1);
            const float cbm = cbar * mask;
            c_im1 *= mask;  h_im1 *= mask;
            out[0L * OUT_SZ + ob + h] = cell;    // all_cell (unmasked)
            out[1L * OUT_SZ + ob + h] = cbm;     // all_cell_bar (masked)
            out[2L * OUT_SZ + ob + h] = gd;      // all_gate_decay
            out[3L * OUT_SZ + ob + h] = go;      // all_gate_output
            out[4L * OUT_SZ + ob + h] = h_im1;   // all_hidden (masked)
            out[5L * OUT_SZ + ob + h] = hmin;    // all_hidden_after_update
            c_st[jj]  = c_im1;
            cb_st[jj] = cbm;
            Xsh[m * XS + h] = (__bf16)h_im1;     // h_plus for next (earlier) step
        }
        __syncthreads();
    }
}

extern "C" void kernel_launch(void* const* d_in, const int* in_sizes, int n_in,
                              void* d_out, int out_size, void* d_ws, size_t ws_size,
                              hipStream_t stream) {
    const int*   ev  = (const int*)d_in[0];     // event_obs [16,8,256] i32
    const float* dt  = (const float*)d_in[1];   // dtime_obs [16,8,256] f32
    const float* Emb = (const float*)d_in[2];   // [35,256]
    const float* W   = (const float*)d_in[3];   // [1792,512]
    const float* b   = (const float*)d_in[4];   // [1792]
    float* out = (float*)d_out;                 // 6 x [128,255,256] f32, concatenated

    __bf16* wh   = (__bf16*)d_ws;                                   // 917,504 B
    float*  proj = (float*)((char*)d_ws + (size_t)SEVEN_H * H * 2); // 250,880 B

    ctlstm_prep_wh  <<<(SEVEN_H * H + 255) / 256, 256, 0, stream>>>(W, wh);
    ctlstm_prep_proj<<<(VOCAB * SEVEN_H + 255) / 256, 256, 0, stream>>>(Emb, W, b, proj);
    ctlstm_scan     <<<NWG, 256, 0, stream>>>(ev, dt, proj, wh, out);
}
